// Repro_11879879541977
// MI455X (gfx1250) — matrix operand order repaired and checked by reading, not compiled, not run on hardware
//
#include <hip/hip_runtime.h>

#define B_   2
#define S_   1024
#define D_   4096
#define H_   32
#define HD_  128
#define NTOK (B_ * S_)
#define SCALE_ 0.08838834764831843f
#define EPS_   1e-5f

typedef __attribute__((ext_vector_type(16))) __bf16 v16bf;
typedef __attribute__((ext_vector_type(8)))  float  v8f;
typedef unsigned short u16;

__device__ __forceinline__ float bf2f(u16 b) {
    unsigned int u = ((unsigned int)b) << 16;
    float f; __builtin_memcpy(&f, &u, 4); return f;
}
__device__ __forceinline__ u16 f2bf(float f) {
    unsigned int u; __builtin_memcpy(&u, &f, 4);
    u += 0x7FFFu + ((u >> 16) & 1u);   // round-to-nearest-even
    return (u16)(u >> 16);
}
// Fill a 16-element bf16 fragment from two 16-byte chunks.
__device__ __forceinline__ void load16(v16bf* dst, const u16* p0, const u16* p1) {
    ((uint4*)dst)[0] = *(const uint4*)p0;
    ((uint4*)dst)[1] = *(const uint4*)p1;
}
// Low 32 bits of a flat LDS pointer == LDS byte offset (aperture hi bits dropped).
__device__ __forceinline__ unsigned lds_off(const void* p) {
    return (unsigned)(size_t)p;
}

// ---------------------------------------------------------------------------
// RMSNorm (optionally fused bf16 residual add): one block per token row.
// B128 vectorized: each thread owns 16 contiguous elements.
// ---------------------------------------------------------------------------
__global__ __launch_bounds__(256)
void rmsnorm_kernel(const u16* __restrict__ x, const u16* __restrict__ res,
                    const u16* __restrict__ w, u16* __restrict__ out)
{
    __shared__ float red[8];
    const int row = blockIdx.x;
    const int tid = threadIdx.x;
    const uint4* x4 = (const uint4*)(x + (size_t)row * D_);
    const uint4* r4 = res ? (const uint4*)(res + (size_t)row * D_) : nullptr;
    float vals[16];
    float ss = 0.f;
#pragma unroll
    for (int i = 0; i < 2; ++i) {
        uint4 xv = x4[tid * 2 + i];
        u16 ex[8]; __builtin_memcpy(ex, &xv, 16);
        u16 er[8];
        if (r4) { uint4 rv = r4[tid * 2 + i]; __builtin_memcpy(er, &rv, 16); }
#pragma unroll
        for (int e = 0; e < 8; ++e) {
            float f = bf2f(ex[e]);
            if (r4) f = bf2f(f2bf(f + bf2f(er[e])));   // h = x + o in bf16
            vals[i * 8 + e] = f;
            ss += f * f;
        }
    }
#pragma unroll
    for (int off = 1; off < 32; off <<= 1) ss += __shfl_xor(ss, off, 32);
    if ((tid & 31) == 0) red[tid >> 5] = ss;
    __syncthreads();
    float tot = 0.f;
#pragma unroll
    for (int wv = 0; wv < 8; ++wv) tot += red[wv];
    const float r = rsqrtf(tot / (float)D_ + EPS_);
#pragma unroll
    for (int i = 0; i < 2; ++i) {
        uint4 wv4 = ((const uint4*)w)[tid * 2 + i];
        u16 ew[8]; __builtin_memcpy(ew, &wv4, 16);
        u16 eo[8];
#pragma unroll
        for (int e = 0; e < 8; ++e) {
            float y = bf2f(f2bf(vals[i * 8 + e] * r));  // cast to bf16 first
            eo[e] = f2bf(y * bf2f(ew[e]));
        }
        uint4 ov; __builtin_memcpy(&ov, eo, 16);
        ((uint4*)(out + (size_t)row * D_))[tid * 2 + i] = ov;
    }
}

// ---------------------------------------------------------------------------
// C[M,N] = A[M,K] * W[N,K]^T   (bf16 in, f32 accumulate via WMMA, bf16 out)
// 128 threads = 4 waves arranged 2(M)x2(N); wave tile 32x64 => 2x4 fragments,
// 8 WMMA per 32-wide K-step at 12 B128 loads (1.5 loads/WMMA).
// ---------------------------------------------------------------------------
__global__ __launch_bounds__(128)
void gemm_bf16_nt(const u16* __restrict__ A, const u16* __restrict__ W,
                  u16* __restrict__ C, int M, int N, int K)
{
    const int lane = threadIdx.x & 31;
    const int wave = threadIdx.x >> 5;
    const int hl   = lane >> 4;     // which 16-lane half
    const int l15  = lane & 15;
    const int m0 = blockIdx.y * 64  + (wave >> 1) * 32;
    const int n0 = blockIdx.x * 128 + (wave & 1) * 64;

    const u16* arow[2];
    arow[0] = A + (size_t)(m0 + l15) * K;
    arow[1] = arow[0] + (size_t)16 * K;
    const u16* wrow[4];
#pragma unroll
    for (int jb = 0; jb < 4; ++jb)
        wrow[jb] = W + (size_t)(n0 + jb * 16 + l15) * K;
    const int ak = hl * 8;     // A-frag lane K offset
    const int bk = hl * 8;     // B-frag lane K offset

    v8f acc[2][4] = {};
    for (int k0 = 0; k0 < K; k0 += 32) {
        if (k0 + 32 < K) {
            __builtin_prefetch(arow[0] + k0 + 32, 0, 3);
            __builtin_prefetch(wrow[0] + k0 + 32, 0, 3);
        }
        v16bf a[2], b[4];
#pragma unroll
        for (int i = 0; i < 2; ++i)
            load16(&a[i], arow[i] + k0 + ak, arow[i] + k0 + ak + 16);
#pragma unroll
        for (int jb = 0; jb < 4; ++jb)
            load16(&b[jb], wrow[jb] + k0 + bk, wrow[jb] + k0 + bk + 16);
#pragma unroll
        for (int i = 0; i < 2; ++i)
#pragma unroll
            for (int jb = 0; jb < 4; ++jb)
                acc[i][jb] = __builtin_amdgcn_wmma_f32_16x16x32_bf16(
                    false, a[i], false, b[jb], (short)0, acc[i][jb], false, false);
    }
    // C layout: VGPR r -> M = r (+8 for upper lane half); lane -> N column.
#pragma unroll
    for (int i = 0; i < 2; ++i) {
        const int mb = m0 + i * 16 + hl * 8;
#pragma unroll
        for (int jb = 0; jb < 4; ++jb) {
            const int nb = n0 + jb * 16 + l15;
#pragma unroll
            for (int r = 0; r < 8; ++r)
                C[(size_t)(mb + r) * N + nb] = f2bf(acc[i][jb][r]);
        }
    }
}

// ---------------------------------------------------------------------------
// RoPE, in place, B128 vectorized: thread = 8 pairs (16 contiguous bf16).
// x: [nTokens, nHeads, 128] bf16; freqs: [1024, 64, 2] f32.
// ---------------------------------------------------------------------------
__global__ __launch_bounds__(256)
void rope_kernel(u16* __restrict__ x, const float* __restrict__ freqs,
                 int nTokens, int nHeads)
{
    int idx = blockIdx.x * blockDim.x + threadIdx.x;
    int total = nTokens * nHeads * 8;
    if (idx >= total) return;
    int c   = idx & 7;                 // 16-element chunk within the head
    int h   = (idx >> 3) % nHeads;
    int tok = idx / (8 * nHeads);
    int s   = tok % S_;
    size_t base = ((size_t)tok * nHeads + h) * HD_ + c * 16;
    uint4 d0 = *(const uint4*)(x + base);
    uint4 d1 = *(const uint4*)(x + base + 8);
    u16 e[16];
    __builtin_memcpy(e,     &d0, 16);
    __builtin_memcpy(e + 8, &d1, 16);
    const float2* fr2 = (const float2*)freqs;
    u16 o[16];
#pragma unroll
    for (int p = 0; p < 8; ++p) {
        float2 cs = fr2[s * 64 + c * 8 + p];
        float xr = bf2f(e[2 * p]), xi = bf2f(e[2 * p + 1]);
        o[2 * p]     = f2bf(xr * cs.x - xi * cs.y);
        o[2 * p + 1] = f2bf(xr * cs.y + xi * cs.x);
    }
    __builtin_memcpy(&d0, o,     16);
    __builtin_memcpy(&d1, o + 8, 16);
    *(uint4*)(x + base)     = d0;
    *(uint4*)(x + base + 8) = d1;
}

// ---------------------------------------------------------------------------
// Flash attention, causal, single KV head (MQA).
// grid = (S/64, H, B); 128 threads = 4 waves, each wave owns 16 q rows.
// K tile staged with CDNA5 async global->LDS (ASYNCcnt); V transposed on the
// way into LDS so PV B-fragments are contiguous loads.
// ---------------------------------------------------------------------------
__global__ __launch_bounds__(128)
void attn_kernel(const u16* __restrict__ q, const u16* __restrict__ k,
                 const u16* __restrict__ v, u16* __restrict__ out)
{
    __shared__ u16 Kt[64 * 128];      // K chunk, row-major [k][d]
    __shared__ u16 VtT[128 * 64];     // V chunk transposed [d][k]
    __shared__ u16 Pt[4][16 * 64];    // per-wave probs staging

    const int lane = threadIdx.x & 31;
    const int wave = threadIdx.x >> 5;
    const int hl   = lane >> 4;
    const int l15  = lane & 15;
    const int b = blockIdx.z, h = blockIdx.y, qt = blockIdx.x;
    const int q0 = qt * 64 + wave * 16;

    // Q fragments (held for the whole kv loop): 4 chunks of K=32 over d=128.
    const u16* qrow = q + ((size_t)(b * S_ + q0 + l15) * H_ + h) * HD_;
    v16bf qa[4];
#pragma unroll
    for (int dd = 0; dd < 4; ++dd) {
        const u16* p = qrow + dd * 32 + hl * 8;
        load16(&qa[dd], p, p + 16);
    }

    float rmax[8], rsum[8];
    v8f o_acc[8] = {};
#pragma unroll
    for (int r = 0; r < 8; ++r) { rmax[r] = -3.0e38f; rsum[r] = 0.f; }

    const u16* kb_g = k + (size_t)b * S_ * HD_;
    const u16* vb_g = v + (size_t)b * S_ * HD_;
    u16* pw = &Pt[wave][0];
    const unsigned ktBase = lds_off(Kt);

    const int nchunk = qt + 1;   // causal: only chunks with kstart <= q_max
    for (int j = 0; j < nchunk; ++j) {
        __syncthreads();   // all waves done reading previous K/V tiles
        // --- K tile: async DMA global -> LDS, bypassing VGPRs (ASYNCcnt) ---
        {
            const char* gk = (const char*)(kb_g + (size_t)j * 64 * HD_);
            for (int t = threadIdx.x; t < 1024; t += 128) {
                unsigned    la = ktBase + (unsigned)t * 16u;
                const void* ga = gk + (size_t)t * 16;
                asm volatile("global_load_async_to_lds_b128 %0, %1, off"
                             :: "v"(la), "v"(ga) : "memory");
            }
        }
        // --- V tile: load row, scatter transposed into LDS ---
        {
            const uint4* vsrc = (const uint4*)(vb_g + (size_t)j * 64 * HD_);
            for (int t = threadIdx.x; t < 1024; t += 128) {
                uint4 val = vsrc[t];
                const u16* pv = (const u16*)&val;
                int krow = t >> 4, dblk = (t & 15) * 8;
#pragma unroll
                for (int e = 0; e < 8; ++e) VtT[(dblk + e) * 64 + krow] = pv[e];
            }
        }
        asm volatile("s_wait_asynccnt 0" ::: "memory");
        __syncthreads();

        // S tile: 16 q rows x 64 k cols, accumulated over d in 4 WMMA steps.
        v8f sfr[4] = {};
#pragma unroll
        for (int jj = 0; jj < 4; ++jj) {
#pragma unroll
            for (int dd = 0; dd < 4; ++dd) {
                v16bf bf;
                const u16* kp = &Kt[(jj * 16 + l15) * 128 + dd * 32 + hl * 8];
                load16(&bf, kp, kp + 16);
                sfr[jj] = __builtin_amdgcn_wmma_f32_16x16x32_bf16(
                    false, qa[dd], false, bf, (short)0, sfr[jj], false, false);
            }
        }

        // scale + causal mask + per-row local max
        const int qbase_row = q0 + hl * 8;
        float mloc[8];
#pragma unroll
        for (int r = 0; r < 8; ++r) mloc[r] = -3.0e38f;
#pragma unroll
        for (int jj = 0; jj < 4; ++jj) {
            const int kcol = j * 64 + jj * 16 + l15;
#pragma unroll
            for (int r = 0; r < 8; ++r) {
                float sv = sfr[jj][r] * SCALE_;
                sv = (kcol <= qbase_row + r) ? sv : -3.0e38f;
                sfr[jj][r] = sv;
                mloc[r] = fmaxf(mloc[r], sv);
            }
        }
        // online softmax rescale factors (rows span lanes 0-15 / 16-31 groups)
        float alpha[8];
#pragma unroll
        for (int r = 0; r < 8; ++r) {
            float m = mloc[r];
#pragma unroll
            for (int off = 1; off < 16; off <<= 1) m = fmaxf(m, __shfl_xor(m, off, 32));
            float mnew = fmaxf(rmax[r], m);
            alpha[r] = __expf(rmax[r] - mnew);
            rmax[r] = mnew;
        }
        float psum[8];
#pragma unroll
        for (int r = 0; r < 8; ++r) psum[r] = 0.f;
#pragma unroll
        for (int jj = 0; jj < 4; ++jj) {
#pragma unroll
            for (int r = 0; r < 8; ++r) {
                float p = __expf(sfr[jj][r] - rmax[r]);
                psum[r] += p;
                pw[(hl * 8 + r) * 64 + jj * 16 + l15] = f2bf(p);
            }
        }
#pragma unroll
        for (int r = 0; r < 8; ++r) {
            float ps = psum[r];
#pragma unroll
            for (int off = 1; off < 16; off <<= 1) ps += __shfl_xor(ps, off, 32);
            rsum[r] = rsum[r] * alpha[r] + ps;
#pragma unroll
            for (int dc = 0; dc < 8; ++dc) o_acc[dc][r] *= alpha[r];
        }
        // make this wave's P stores visible to its own A-frag reloads
        asm volatile("s_wait_dscnt 0" ::: "memory");
        // O += P @ V  (K=64 split into two 32-wide WMMA steps, 8 d-blocks)
#pragma unroll
        for (int kk = 0; kk < 2; ++kk) {
            v16bf pf;
            const u16* pp = pw + l15 * 64 + kk * 32 + hl * 8;
            load16(&pf, pp, pp + 16);
#pragma unroll
            for (int dc = 0; dc < 8; ++dc) {
                v16bf vf;
                const u16* vp = &VtT[(dc * 16 + l15) * 64 + kk * 32 + hl * 8];
                load16(&vf, vp, vp + 16);
                o_acc[dc] = __builtin_amdgcn_wmma_f32_16x16x32_bf16(
                    false, pf, false, vf, (short)0, o_acc[dc], false, false);
            }
        }
    }

    // normalize and write attn output [B, S, H*HD]
#pragma unroll
    for (int dc = 0; dc < 8; ++dc) {
        const int dcol = h * HD_ + dc * 16 + l15;
#pragma unroll
        for (int r = 0; r < 8; ++r) {
            const int srow = q0 + hl * 8 + r;
            out[(size_t)(b * S_ + srow) * (size_t)D_ + dcol] = f2bf(o_acc[dc][r] / rsum[r]);
        }
    }
}

// ---------------------------------------------------------------------------
extern "C" void kernel_launch(void* const* d_in, const int* in_sizes, int n_in,
                              void* d_out, int out_size, void* d_ws, size_t ws_size,
                              hipStream_t stream)
{
    (void)in_sizes; (void)n_in; (void)out_size; (void)ws_size;
    const u16*   x  = (const u16*)d_in[0];
    const u16*   w1 = (const u16*)d_in[1];
    const u16*   wq = (const u16*)d_in[2];
    const u16*   wk = (const u16*)d_in[3];
    const u16*   wv = (const u16*)d_in[4];
    const u16*   wo = (const u16*)d_in[5];
    const u16*   w2 = (const u16*)d_in[6];
    const float* fr = (const float*)d_in[7];
    u16* out = (u16*)d_out;

    char* ws = (char*)d_ws;
    const size_t MB = 1ull << 20;
    u16* xn = (u16*)(ws);                      // 16 MB  [2048,4096]
    u16* qb = (u16*)(ws + 16 * MB);            // 16 MB  [2048,4096]
    u16* ab = (u16*)(ws + 32 * MB);            // 16 MB  [2048,4096]
    u16* ob = (u16*)(ws + 48 * MB);            // 16 MB  [2048,4096]
    u16* kb = (u16*)(ws + 64 * MB);            // 512 KB [2048,128]
    u16* vb = (u16*)(ws + 64 * MB + 512 * 1024);

    rmsnorm_kernel<<<NTOK, 256, 0, stream>>>(x, nullptr, w1, xn);

    gemm_bf16_nt<<<dim3(D_  / 128, NTOK / 64), 128, 0, stream>>>(xn, wq, qb, NTOK, D_,  D_);
    gemm_bf16_nt<<<dim3(HD_ / 128, NTOK / 64), 128, 0, stream>>>(xn, wk, kb, NTOK, HD_, D_);
    gemm_bf16_nt<<<dim3(HD_ / 128, NTOK / 64), 128, 0, stream>>>(xn, wv, vb, NTOK, HD_, D_);

    rope_kernel<<<(NTOK * H_ * 8) / 256, 256, 0, stream>>>(qb, fr, NTOK, H_);
    rope_kernel<<<(NTOK * 1  * 8) / 256, 256, 0, stream>>>(kb, fr, NTOK, 1);

    attn_kernel<<<dim3(S_ / 64, H_, B_), 128, 0, stream>>>(qb, kb, vb, ab);

    gemm_bf16_nt<<<dim3(D_ / 128, NTOK / 64), 128, 0, stream>>>(ab, wo, ob, NTOK, D_, D_);

    rmsnorm_kernel<<<NTOK, 256, 0, stream>>>(x, ob, w2, out);
}
